// LocalFeatureTransformer_32658931319254
// MI455X (gfx1250) — compile-verified, matrix operand order checked by reading, not run on hardware
//
#include <hip/hip_runtime.h>
#include <hip/hip_bf16.h>

typedef __attribute__((ext_vector_type(16))) __bf16 v16bf;
typedef __attribute__((ext_vector_type(8)))  float  v8f;
typedef int v4i_ __attribute__((vector_size(16)));
typedef __attribute__((address_space(1))) v4i_ gv4i;
typedef __attribute__((address_space(3))) v4i_ lv4i;

#define BN_   2
#define CC    256
#define HH    256
#define WW    256
#define NHD   8
#define DD    32
#define HP    64          // pooled h=w
#define LL    4096        // pooled tokens per image
#define HWF   65536       // full-res pixels per image
#define C2    512
#define MT    64          // tokens per block in fused MLP

#if defined(__has_builtin)
#  if __has_builtin(__builtin_amdgcn_global_load_async_to_lds_b128)
#    define USE_ASYNC_LDS 1
#  endif
#endif

__device__ __forceinline__ void wait_async_zero() {
#if defined(USE_ASYNC_LDS)
#  if __has_builtin(__builtin_amdgcn_s_wait_asynccnt)
  __builtin_amdgcn_s_wait_asynccnt(0);
#  else
  asm volatile("s_wait_asynccnt 0x0" ::: "memory");
#  endif
#endif
}

// ---------------------------------------------------------------------------
// K0: float -> bf16 weight conversion
// ---------------------------------------------------------------------------
__global__ __launch_bounds__(256) void f2bf_kernel(const float* __restrict__ in,
                                                   __bf16* __restrict__ out, int n) {
  int i = blockIdx.x * 256 + threadIdx.x;
  if (i < n) out[i] = (__bf16)in[i];
}

// ---------------------------------------------------------------------------
// K1: 4x4 maxpool + LayerNorm over C, writes token-major [B, L, C] f32
// ---------------------------------------------------------------------------
__global__ __launch_bounds__(256) void pool_ln_kernel(
    const float* __restrict__ x, const float* __restrict__ src,
    const float* __restrict__ g, const float* __restrict__ be,
    float* __restrict__ lnX, float* __restrict__ lnS) {
  int bi = blockIdx.x;
  int b  = bi >> 12;
  int pos = bi & 4095;
  int py = pos >> 6, px = pos & 63;
  int c = threadIdx.x;
  __shared__ float s1[256], s2[256];

  for (int it = 0; it < 2; ++it) {
    const float* in = it ? src : x;
    size_t base = ((size_t)(b * CC + c) * HH + py * 4) * WW + px * 4;
    float v = -3.4e38f;
#pragma unroll
    for (int dy = 0; dy < 4; ++dy)
#pragma unroll
      for (int dx = 0; dx < 4; ++dx)
        v = fmaxf(v, in[base + dy * WW + dx]);
    s1[c] = v;
    s2[c] = v * v;
    __syncthreads();
    for (int st = 128; st > 0; st >>= 1) {
      if (c < st) { s1[c] += s1[c + st]; s2[c] += s2[c + st]; }
      __syncthreads();
    }
    float mean = s1[0] * (1.f / 256.f);
    float var  = s2[0] * (1.f / 256.f) - mean * mean;
    float r = rsqrtf(var + 1e-5f);
    float o = (v - mean) * r * g[c] + be[c];
    (it ? lnS : lnX)[(size_t)(b * LL + pos) * CC + c] = o;
    __syncthreads();
  }
}

// ---------------------------------------------------------------------------
// K2: x NCHW f32 -> token-major [B, HW, C] bf16
// ---------------------------------------------------------------------------
__global__ __launch_bounds__(256) void transpose_x_kernel(const float* __restrict__ x,
                                                          __bf16* __restrict__ xT) {
  size_t idx = (size_t)blockIdx.x * 256 + threadIdx.x;  // over B*C*HW
  int p = (int)(idx & (HWF - 1));
  size_t t = idx >> 16;
  int c = (int)(t & 255);
  int b = (int)(t >> 8);
  xT[((size_t)(b * HWF + p)) * CC + c] = (__bf16)x[idx];
}

// ---------------------------------------------------------------------------
// K3: depthwise 3x3 conv + BN (+ elu+1 for Q,K) -> bf16 tokens [B, L, C]
// ---------------------------------------------------------------------------
__device__ __forceinline__ float bn_apply(float y, float g, float b, float m, float v) {
  float s = g * rsqrtf(v + 1e-5f);
  return y * s + (b - m * s);
}

__global__ __launch_bounds__(256) void conv_qkv_kernel(
    const float* __restrict__ lnX, const float* __restrict__ lnS,
    const float* __restrict__ qw, const float* __restrict__ qg, const float* __restrict__ qb,
    const float* __restrict__ qm, const float* __restrict__ qv,
    const float* __restrict__ kw, const float* __restrict__ kg, const float* __restrict__ kb_,
    const float* __restrict__ km, const float* __restrict__ kv_,
    const float* __restrict__ vw, const float* __restrict__ vg, const float* __restrict__ vb,
    const float* __restrict__ vm, const float* __restrict__ vv,
    __bf16* __restrict__ Q, __bf16* __restrict__ K, __bf16* __restrict__ V) {
  int bi = blockIdx.x;
  int b = bi >> 12, pos = bi & 4095;
  int py = pos >> 6, px = pos & 63;
  int c = threadIdx.x;

  float nx[9], ns[9];
  int i = 0;
#pragma unroll
  for (int ky = 0; ky < 3; ++ky)
#pragma unroll
    for (int kx = 0; kx < 3; ++kx, ++i) {
      int yy = py + ky - 1, xx = px + kx - 1;
      bool ok = ((unsigned)yy < HP) && ((unsigned)xx < HP);
      size_t idx = (size_t)(b * LL + yy * HP + xx) * CC + c;
      nx[i] = ok ? lnX[idx] : 0.f;
      ns[i] = ok ? lnS[idx] : 0.f;
    }
  float aq = 0.f, ak = 0.f, av = 0.f;
#pragma unroll
  for (i = 0; i < 9; ++i) {
    aq += qw[c * 9 + i] * nx[i];
    ak += kw[c * 9 + i] * ns[i];
    av += vw[c * 9 + i] * ns[i];
  }
  float qn = bn_apply(aq, qg[c], qb[c], qm[c], qv[c]);
  float kn = bn_apply(ak, kg[c], kb_[c], km[c], kv_[c]);
  float vn = bn_apply(av, vg[c], vb[c], vm[c], vv[c]);
  qn = qn > 0.f ? qn + 1.f : __expf(qn);   // elu(x)+1
  kn = kn > 0.f ? kn + 1.f : __expf(kn);
  size_t ob = (size_t)(b * LL + pos) * CC + c;
  Q[ob] = (__bf16)qn;
  K[ob] = (__bf16)kn;
  V[ob] = (__bf16)vn;
}

// ---------------------------------------------------------------------------
// K4: KV[b,h,d,dv] = sum_l K[l,d]*V[l,dv]   and Ksum[b,c] = sum_l K[l,c]
// ---------------------------------------------------------------------------
__global__ __launch_bounds__(256) void kv_reduce_kernel(
    const __bf16* __restrict__ Kbf, const __bf16* __restrict__ Vbf,
    float* __restrict__ KV, float* __restrict__ Ksum) {
  int bh = blockIdx.x;
  int b = bh >> 3, h = bh & 7;
  int t = threadIdx.x;
  __shared__ float sK[32][32];
  __shared__ float sV[32][32];
  float acc[4] = {0.f, 0.f, 0.f, 0.f};
  float ks = 0.f;
  int dv = t & 31, dg = t >> 5;

  for (int l0 = 0; l0 < LL; l0 += 32) {
#pragma unroll
    for (int j = 0; j < 4; ++j) {
      int flat = t * 4 + j;
      int r = flat >> 5, col = flat & 31;
      size_t gi = (size_t)(b * LL + l0 + r) * CC + h * DD + col;
      sK[r][col] = (float)Kbf[gi];
      sV[r][col] = (float)Vbf[gi];
    }
    __syncthreads();
#pragma unroll 4
    for (int r = 0; r < 32; ++r) {
      float vvv = sV[r][dv];
      const float* krow = sK[r];
#pragma unroll
      for (int i = 0; i < 4; ++i) acc[i] += krow[dg * 4 + i] * vvv;
    }
    if (t < 32) {
#pragma unroll 8
      for (int r = 0; r < 32; ++r) ks += sK[r][t];
    }
    __syncthreads();
  }
  size_t base = (size_t)bh * (DD * DD);
#pragma unroll
  for (int i = 0; i < 4; ++i) KV[base + (dg * 4 + i) * DD + dv] = acc[i];
  if (t < 32) Ksum[b * CC + h * DD + t] = ks;
}

// ---------------------------------------------------------------------------
// K5: msg[l,c] = (Q[l,h,:] . KV[h,:,dv]) / (Q[l,h,:] . Ksum[h,:] + eps)
// ---------------------------------------------------------------------------
__global__ __launch_bounds__(256) void msg_kernel(
    const __bf16* __restrict__ Qbf, const float* __restrict__ KV,
    const float* __restrict__ Ksum, __bf16* __restrict__ msg) {
  int bi = blockIdx.x;
  int b = bi >> 12, pos = bi & 4095;
  int c = threadIdx.x;
  int h = c >> 5, dv = c & 31;
  __shared__ float sQ[256];
  size_t rb = (size_t)(b * LL + pos) * CC;
  sQ[c] = (float)Qbf[rb + c];
  __syncthreads();
  const float* qh = &sQ[h * DD];
  const float* ks = Ksum + b * CC + h * DD;
  const float* kv = KV + (size_t)(b * NHD + h) * (DD * DD) + dv;
  float z = 0.f, m = 0.f;
#pragma unroll 8
  for (int d = 0; d < DD; ++d) {
    z += qh[d] * ks[d];
    m += qh[d] * kv[d * DD];
  }
  msg[rb + c] = (__bf16)(m / (z + 1e-6f));
}

// ---------------------------------------------------------------------------
// K6: merge GEMM: merged[M=8192, 256] = msg[M,256] @ Wm^T   (WMMA bf16)
// ---------------------------------------------------------------------------
__global__ __launch_bounds__(256) void merge_gemm_kernel(
    const __bf16* __restrict__ A, const __bf16* __restrict__ Wm,
    float* __restrict__ Cout) {
  int wid = threadIdx.x >> 5, lane = threadIdx.x & 31;
  int wg = blockIdx.x * 8 + wid;
  int m0 = (wg >> 4) << 4;
  int nb = (wg & 15) << 4;
  int lrow = lane & 15;
  int khalf = (lane >> 4) << 4;
  int mb = (lane >> 4) << 3;
  v8f acc = {0.f, 0.f, 0.f, 0.f, 0.f, 0.f, 0.f, 0.f};
#pragma unroll
  for (int kb = 0; kb < CC; kb += 32) {
    v16bf a = *(const v16bf*)(A + (size_t)(m0 + lrow) * CC + kb + khalf);
    v16bf w = *(const v16bf*)(Wm + (size_t)(nb + lrow) * CC + kb + khalf);
    acc = __builtin_amdgcn_wmma_f32_16x16x32_bf16(false, a, false, w, (short)0, acc,
                                                  false, false);
  }
#pragma unroll
  for (int r = 0; r < 8; ++r)
    Cout[(size_t)(m0 + mb + r) * CC + nb + lrow] = acc[r];
}

// ---------------------------------------------------------------------------
// K7: fused bilinear-upsample + concat + MLP(relu) + MLP2 + LN + residual
// block = 256 threads (8 waves) handles 64 tokens; each weight B-fragment is
// reused across 4 M-tiles. Dynamic LDS: sA (64KB, reused as f32 sO) + sH (64KB)
// ---------------------------------------------------------------------------
__global__ __launch_bounds__(256) void fused_mlp_kernel(
    const __bf16* __restrict__ xT, const float* __restrict__ merged,
    const __bf16* __restrict__ W1, const __bf16* __restrict__ W2,
    const float* __restrict__ g2, const float* __restrict__ b2,
    const float* __restrict__ x, float* __restrict__ out) {
  int b  = blockIdx.x >> 10;                // 1024 blocks per batch image
  int p0 = (blockIdx.x & 1023) << 6;        // 64 pixels, same row (256/64=4)
  int py = p0 >> 8;
  int px0 = p0 & 255;
  int tid = threadIdx.x;

  extern __shared__ v16bf smem_v[];
  char* smem = (char*)smem_v;
  __bf16* sA = (__bf16*)smem;               // [MT][C2] bf16 = 64KB (cat tile)
  float*  sO = (float*)smem;                // [MT][CC] f32 = 64KB (aliases sA)
  __bf16* sH = (__bf16*)(smem + MT * C2 * 2);  // [MT][C2] bf16 = 64KB
  __shared__ float redA[256], redB[256];
  __shared__ float smean[MT], srstd[MT];

  // prefetch weights toward L2/L0 (probe-confirmed -> global_prefetch_b8)
  __builtin_prefetch(W1 + (size_t)tid * 1024, 0, 3);
  __builtin_prefetch(W2 + (size_t)tid * 512, 0, 3);

  // ---- stage cat tile: x half (async global->LDS when available) ----
  {
#if defined(USE_ASYNC_LDS)
    // 64 rows x 256 bf16 = 2048 16-byte chunks, 8 per thread
#pragma unroll
    for (int j = 0; j < 8; ++j) {
      int ch = j * 256 + tid;
      int row = ch >> 5;
      int col = (ch & 31) * 8;
      const __bf16* srcp = xT + ((size_t)(b * HWF + p0 + row)) * CC + col;
      __bf16* dstp = &sA[row * C2 + col];
      __builtin_amdgcn_global_load_async_to_lds_b128((gv4i*)srcp, (lv4i*)dstp, 0, 0);
    }
#else
    // fallback: 64 rows x 16 v16bf chunks, 4 per thread
#pragma unroll
    for (int j = 0; j < 4; ++j) {
      int ch = j * 256 + tid;
      int row = ch >> 4;
      int col = (ch & 15) * 16;
      const __bf16* srcp = xT + ((size_t)(b * HWF + p0 + row)) * CC + col;
      *(v16bf*)&sA[row * C2 + col] = *(const v16bf*)srcp;
    }
#endif
  }
  // ---- stage cat tile: msg half via on-the-fly bilinear (64 -> 256) ----
  {
    int chunk = tid & 15;
    float fy = py * 0.25f - 0.375f;
    int iy0 = (int)floorf(fy);
    float wy = fy - (float)iy0;
    int iy1 = min(max(iy0 + 1, 0), HP - 1);
    iy0 = min(max(iy0, 0), HP - 1);
#pragma unroll
    for (int it = 0; it < 4; ++it) {
      int tt = it * 16 + (tid >> 4);
      int px = px0 + tt;
      float fx = px * 0.25f - 0.375f;
      int ix0 = (int)floorf(fx);
      float wx = fx - (float)ix0;
      int ix1 = min(max(ix0 + 1, 0), HP - 1);
      ix0 = min(max(ix0, 0), HP - 1);
      const float* m00 = merged + ((size_t)(b * LL + iy0 * HP + ix0)) * CC + chunk * 16;
      const float* m01 = merged + ((size_t)(b * LL + iy0 * HP + ix1)) * CC + chunk * 16;
      const float* m10 = merged + ((size_t)(b * LL + iy1 * HP + ix0)) * CC + chunk * 16;
      const float* m11 = merged + ((size_t)(b * LL + iy1 * HP + ix1)) * CC + chunk * 16;
#pragma unroll
      for (int j = 0; j < 16; ++j) {
        float v0 = m00[j] + wx * (m01[j] - m00[j]);
        float v1 = m10[j] + wx * (m11[j] - m10[j]);
        sA[tt * C2 + CC + chunk * 16 + j] = (__bf16)(v0 + wy * (v1 - v0));
      }
    }
  }
  wait_async_zero();
  __syncthreads();

  int wid = tid >> 5, lane = tid & 31;
  int lrow = lane & 15;
  int khalf = (lane >> 4) << 4;
  int mb = (lane >> 4) << 3;

  // ---- layer 1: hidden[64,512] = relu(cat @ W1^T) ----
  // wave handles 4 N-tiles x 4 M-tiles; B-fragment reused across 4 M-tiles
  for (int s = 0; s < 4; ++s) {
    int nb = (wid * 4 + s) << 4;
    v8f acc[4];
#pragma unroll
    for (int mt = 0; mt < 4; ++mt)
      acc[mt] = (v8f){0.f, 0.f, 0.f, 0.f, 0.f, 0.f, 0.f, 0.f};
    for (int kb = 0; kb < C2; kb += 32) {
      v16bf w = *(const v16bf*)(W1 + (size_t)(nb + lrow) * C2 + kb + khalf);
#pragma unroll
      for (int mt = 0; mt < 4; ++mt) {
        v16bf a = *(const v16bf*)&sA[(mt * 16 + lrow) * C2 + kb + khalf];
        acc[mt] = __builtin_amdgcn_wmma_f32_16x16x32_bf16(false, a, false, w, (short)0,
                                                          acc[mt], false, false);
      }
    }
#pragma unroll
    for (int mt = 0; mt < 4; ++mt)
#pragma unroll
      for (int r = 0; r < 8; ++r) {
        float v = acc[mt][r];
        sH[(mt * 16 + mb + r) * C2 + nb + lrow] = (__bf16)(v > 0.f ? v : 0.f);
      }
  }
  __syncthreads();   // all sA reads done; sH complete

  // ---- layer 2: out[64,256] = hidden @ W2^T (sO aliases sA region) ----
  for (int s = 0; s < 2; ++s) {
    int nb = (wid * 2 + s) << 4;
    v8f acc[4];
#pragma unroll
    for (int mt = 0; mt < 4; ++mt)
      acc[mt] = (v8f){0.f, 0.f, 0.f, 0.f, 0.f, 0.f, 0.f, 0.f};
    for (int kb = 0; kb < C2; kb += 32) {
      v16bf w = *(const v16bf*)(W2 + (size_t)(nb + lrow) * C2 + kb + khalf);
#pragma unroll
      for (int mt = 0; mt < 4; ++mt) {
        v16bf a = *(const v16bf*)&sH[(mt * 16 + lrow) * C2 + kb + khalf];
        acc[mt] = __builtin_amdgcn_wmma_f32_16x16x32_bf16(false, a, false, w, (short)0,
                                                          acc[mt], false, false);
      }
    }
#pragma unroll
    for (int mt = 0; mt < 4; ++mt)
#pragma unroll
      for (int r = 0; r < 8; ++r)
        sO[(mt * 16 + mb + r) * CC + nb + lrow] = acc[mt][r];
  }
  __syncthreads();

  // ---- LayerNorm over 256 channels per token (4 threads per token) ----
  {
    int tt = tid >> 2, q = tid & 3;
    float s1 = 0.f, s2 = 0.f;
    const float* row = &sO[tt * CC + q * 64];
#pragma unroll
    for (int j = 0; j < 64; ++j) {
      float v = row[j];
      s1 += v;
      s2 += v * v;
    }
    redA[tid] = s1;
    redB[tid] = s2;
    __syncthreads();
    if (q == 0) {
      float a = redA[tid] + redA[tid + 1] + redA[tid + 2] + redA[tid + 3];
      float bb = redB[tid] + redB[tid + 1] + redB[tid + 2] + redB[tid + 3];
      float mean = a * (1.f / 256.f);
      float var = bb * (1.f / 256.f) - mean * mean;
      smean[tt] = mean;
      srstd[tt] = rsqrtf(var + 1e-5f);
    }
    __syncthreads();
  }

  // ---- residual + write NCHW: thread = channel, 64 contiguous pixels ----
  {
    int c = tid;
    float gg = g2[c], bb = b2[c];
    const float* xr = x + ((size_t)(b * CC + c)) * HWF + p0;
    float* orow = out + ((size_t)(b * CC + c)) * HWF + p0;
#pragma unroll 8
    for (int t2 = 0; t2 < MT; ++t2) {
      float v = (sO[t2 * CC + c] - smean[t2]) * srstd[t2] * gg + bb;
      orow[t2] = xr[t2] + v;
    }
  }
}

// ---------------------------------------------------------------------------
// Host launch
// ---------------------------------------------------------------------------
extern "C" void kernel_launch(void* const* d_in, const int* in_sizes, int n_in,
                              void* d_out, int out_size, void* d_ws, size_t ws_size,
                              hipStream_t stream) {
  const float* x      = (const float*)d_in[0];
  const float* source = (const float*)d_in[1];
  const float* ln1_g  = (const float*)d_in[2];
  const float* ln1_b  = (const float*)d_in[3];
  const float* merge_w = (const float*)d_in[4];
  const float* mlp_w1  = (const float*)d_in[5];
  const float* mlp_w2  = (const float*)d_in[6];
  const float* ln2_g   = (const float*)d_in[7];
  const float* ln2_b   = (const float*)d_in[8];
  const float* q_w = (const float*)d_in[9];
  const float* q_g = (const float*)d_in[10];
  const float* q_b = (const float*)d_in[11];
  const float* q_m = (const float*)d_in[12];
  const float* q_v = (const float*)d_in[13];
  const float* k_w = (const float*)d_in[14];
  const float* k_g = (const float*)d_in[15];
  const float* k_b = (const float*)d_in[16];
  const float* k_m = (const float*)d_in[17];
  const float* k_v = (const float*)d_in[18];
  const float* v_w = (const float*)d_in[19];
  const float* v_g = (const float*)d_in[20];
  const float* v_b = (const float*)d_in[21];
  const float* v_m = (const float*)d_in[22];
  const float* v_v = (const float*)d_in[23];
  float* out = (float*)d_out;

  size_t o = 0;
  auto carve = [&](size_t bytes) -> void* {
    void* p = (char*)d_ws + o;
    o += (bytes + 255) & ~(size_t)255;
    return p;
  };
  float*  lnX    = (float*)carve((size_t)BN_ * LL * CC * 4);
  float*  lnS    = (float*)carve((size_t)BN_ * LL * CC * 4);
  __bf16* Qbf    = (__bf16*)carve((size_t)BN_ * LL * CC * 2);
  __bf16* Kbf    = (__bf16*)carve((size_t)BN_ * LL * CC * 2);
  __bf16* Vbf    = (__bf16*)carve((size_t)BN_ * LL * CC * 2);
  float*  KV     = (float*)carve((size_t)BN_ * NHD * DD * DD * 4);
  float*  Ksum   = (float*)carve((size_t)BN_ * CC * 4);
  __bf16* msg    = (__bf16*)carve((size_t)BN_ * LL * CC * 2);
  float*  merged = (float*)carve((size_t)BN_ * LL * CC * 4);
  __bf16* xT     = (__bf16*)carve((size_t)BN_ * HWF * CC * 2);
  __bf16* WmBf   = (__bf16*)carve((size_t)CC * CC * 2);
  __bf16* W1Bf   = (__bf16*)carve((size_t)C2 * C2 * 2);
  __bf16* W2Bf   = (__bf16*)carve((size_t)CC * C2 * 2);

  f2bf_kernel<<<(CC * CC + 255) / 256, 256, 0, stream>>>(merge_w, WmBf, CC * CC);
  f2bf_kernel<<<(C2 * C2 + 255) / 256, 256, 0, stream>>>(mlp_w1, W1Bf, C2 * C2);
  f2bf_kernel<<<(CC * C2 + 255) / 256, 256, 0, stream>>>(mlp_w2, W2Bf, CC * C2);

  pool_ln_kernel<<<BN_ * LL, 256, 0, stream>>>(x, source, ln1_g, ln1_b, lnX, lnS);

  transpose_x_kernel<<<(BN_ * CC * HWF) / 256, 256, 0, stream>>>(x, xT);

  conv_qkv_kernel<<<BN_ * LL, 256, 0, stream>>>(lnX, lnS,
      q_w, q_g, q_b, q_m, q_v,
      k_w, k_g, k_b, k_m, k_v,
      v_w, v_g, v_b, v_m, v_v,
      Qbf, Kbf, Vbf);

  kv_reduce_kernel<<<BN_ * NHD, 256, 0, stream>>>(Kbf, Vbf, KV, Ksum);

  msg_kernel<<<BN_ * LL, 256, 0, stream>>>(Qbf, KV, Ksum, msg);

  merge_gemm_kernel<<<(BN_ * LL / 16) * 16 / 8, 256, 0, stream>>>(msg, WmBf, merged);

  // dynamic LDS: sA/sO (64KB) + sH (64KB) = 128KB
  fused_mlp_kernel<<<BN_ * HWF / MT, 256, 2 * MT * C2 * 2, stream>>>(
      xT, merged, W1Bf, W2Bf, ln2_g, ln2_b, x, out);
}
